// BigBirdBlock_14980845928963
// MI455X (gfx1250) — compile-verified, matrix operand order checked by reading, not run on hardware
//
#include <hip/hip_runtime.h>

// ---------------------------------------------------------------------------
// BigBird block for MI455X (gfx1250, wave32, WMMA bf16)
// B=2 L=4096 D=768 H=12 HD=64 NB=64 BS=64 R=3 MLP=3072
// ---------------------------------------------------------------------------

#define Bc   2
#define Lc   4096
#define Dc   768
#define Hc   12
#define HDc  64
#define NBc  64
#define MLPc 3072
#define Mrows (Bc * Lc)          // 8192

typedef __attribute__((ext_vector_type(16))) __bf16 v16bf;
typedef __attribute__((ext_vector_type(8)))  float  v8f;

// float -> bf16, round-to-nearest-even
__device__ __forceinline__ __bf16 f2bf(float f) {
    unsigned u = __builtin_bit_cast(unsigned, f);
    u = (u + 0x7fffu + ((u >> 16) & 1u)) >> 16;
    unsigned short s = (unsigned short)u;
    return __builtin_bit_cast(__bf16, s);
}

// Load one 16x32 bf16 fragment (ISA 7.12.2 16-bit A layout; caller passes
// p = base + row*ld + kc + (lane>>4)*8 ; fetches k[0..7] and k[16..23]).
__device__ __forceinline__ v16bf frag_ld(const __bf16* p) {
    union { uint4 u[2]; v16bf v; } f;
    f.u[0] = *reinterpret_cast<const uint4*>(p);
    f.u[1] = *reinterpret_cast<const uint4*>(p + 16);
    return f.v;
}

__device__ __forceinline__ v8f wmma_bf16(v16bf a, v16bf b, v8f c) {
    return __builtin_amdgcn_wmma_f32_16x16x32_bf16(false, a, false, b,
                                                   (short)0, c, false, false);
}

__device__ __forceinline__ v8f v8f_zero() {
    v8f z;
#pragma unroll
    for (int e = 0; e < 8; ++e) z[e] = 0.0f;
    return z;
}

// ---------------------------------------------------------------------------
// 1. Static BigBird plan: exact replica of np.random.RandomState(0) choice
// ---------------------------------------------------------------------------
__global__ void plan_kernel(int* __restrict__ plan) {
    if (threadIdx.x != 0) return;
    unsigned mt[624];
    int mti;
    {   // numpy mt19937_seed(state, 0)
        unsigned s = 0u;
        for (int i = 0; i < 624; ++i) {
            mt[i] = s;
            s = 1812433253u * (s ^ (s >> 30)) + (unsigned)(i + 1);
        }
        mti = 624;
    }
    auto genrand = [&]() -> unsigned {
        if (mti >= 624) {
            for (int i = 0; i < 624; ++i) {
                unsigned y = (mt[i] & 0x80000000u) | (mt[(i + 1) % 624] & 0x7fffffffu);
                unsigned v = mt[(i + 397) % 624] ^ (y >> 1);
                if (y & 1u) v ^= 0x9908b0dfu;
                mt[i] = v;
            }
            mti = 0;
        }
        unsigned y = mt[mti++];
        y ^= y >> 11;
        y ^= (y << 7)  & 0x9d2c5680u;
        y ^= (y << 15) & 0xefc60000u;
        y ^= y >> 18;
        return y;
    };
    auto interval = [&](unsigned mx) -> unsigned {   // rk_interval: [0, mx]
        if (mx == 0u) return 0u;
        unsigned mask = mx;
        mask |= mask >> 1;  mask |= mask >> 2; mask |= mask >> 4;
        mask |= mask >> 8;  mask |= mask >> 16;
        unsigned v;
        do { v = genrand() & mask; } while (v > mx);
        return v;
    };
    for (int i = 1; i <= NBc - 2; ++i) {
        int cand[NBc]; int pop = 0;
        for (int j = 0; j < NBc; ++j) {
            if (j == 0 || j == NBc - 1 || j == i - 1 || j == i || j == i + 1) continue;
            cand[pop++] = j;
        }
        int perm[NBc];
        for (int j = 0; j < pop; ++j) perm[j] = j;
        for (int j = pop - 1; j > 0; --j) {          // legacy shuffle
            unsigned jj = interval((unsigned)j);
            int t = perm[j]; perm[j] = perm[jj]; perm[jj] = t;
        }
        int* row = plan + (i - 1) * 8;
        row[0] = 0; row[1] = NBc - 1; row[2] = i - 1; row[3] = i; row[4] = i + 1;
        for (int j = 0; j < 3; ++j) row[5 + j] = cand[perm[j]];
    }
}

// ---------------------------------------------------------------------------
// 2. Weight transpose + f32->bf16 (+alpha)   in: R x C, out: C x R
// ---------------------------------------------------------------------------
__global__ void wtrans_kernel(const float* __restrict__ in, __bf16* __restrict__ out,
                              int R, int C, float alpha) {
    __shared__ float tile[32][33];
    int bx = blockIdx.x * 32, by = blockIdx.y * 32;
    int tx = threadIdx.x, ty = threadIdx.y;            // 32 x 8
#pragma unroll
    for (int k = 0; k < 32; k += 8)
        tile[ty + k][tx] = in[(size_t)(by + ty + k) * C + bx + tx];
    __syncthreads();
#pragma unroll
    for (int k = 0; k < 32; k += 8)
        out[(size_t)(bx + ty + k) * R + by + tx] = f2bf(alpha * tile[tx][ty + k]);
}

// ---------------------------------------------------------------------------
// 3. LayerNorm rows of D=768 : f32 in -> bf16 out.  One 256-thread WG / row.
// ---------------------------------------------------------------------------
__global__ void ln_kernel(const float* __restrict__ x,
                          const float* __restrict__ sc,
                          const float* __restrict__ bi,
                          __bf16* __restrict__ out) {
    int row = blockIdx.x;
    const float* xr = x + (size_t)row * Dc;
    float vals[3], s = 0.f, s2 = 0.f;
#pragma unroll
    for (int i = 0; i < 3; ++i) {
        float v = xr[threadIdx.x + i * 256];
        vals[i] = v; s += v; s2 += v * v;
    }
    __shared__ float rs[8], rs2[8];
    int wave = threadIdx.x >> 5, lane = threadIdx.x & 31;
#pragma unroll
    for (int d = 16; d; d >>= 1) { s += __shfl_down(s, d); s2 += __shfl_down(s2, d); }
    if (!lane) { rs[wave] = s; rs2[wave] = s2; }
    __syncthreads();
    float ts = 0.f, ts2 = 0.f;
#pragma unroll
    for (int i = 0; i < 8; ++i) { ts += rs[i]; ts2 += rs2[i]; }
    float mu  = ts * (1.0f / Dc);
    float var = ts2 * (1.0f / Dc) - mu * mu;
    float inv = rsqrtf(var + 1e-6f);
#pragma unroll
    for (int i = 0; i < 3; ++i) {
        int c = threadIdx.x + i * 256;
        out[(size_t)row * Dc + c] = f2bf((vals[i] - mu) * inv * sc[c] + bi[c]);
    }
}

// ---------------------------------------------------------------------------
// 4. bf16 WMMA GEMM:  C[M,N] = A[M,K] * BT[N,K]^T (+bias)(+relu)(+residual)
//    WG = 256 thr (8 waves), tile 128(M) x 64(N); wave = 32x32 = 2x2 wmma.
//    K-loop software-pipelined one step ahead.
// ---------------------------------------------------------------------------
template <bool BIAS, bool RELU, bool RESID, bool OUTBF, bool OUTF>
__global__ void gemm_kernel(const __bf16* __restrict__ A,
                            const __bf16* __restrict__ BT,
                            const float*  __restrict__ bias,
                            const float*  __restrict__ resid,
                            __bf16* __restrict__ outb,
                            float*  __restrict__ outf,
                            int M, int N, int K) {
    int wave = threadIdx.x >> 5, lane = threadIdx.x & 31;
    int r = lane & 15, hf = lane >> 4;
    int m0 = blockIdx.y * 128 + (wave >> 1) * 32;
    int n0 = blockIdx.x * 64  + (wave & 1) * 32;
    v8f acc[2][2];
#pragma unroll
    for (int i = 0; i < 2; ++i)
#pragma unroll
        for (int j = 0; j < 2; ++j) acc[i][j] = v8f_zero();

    const __bf16* ap = A  + (size_t)(m0 + r) * K + hf * 8;
    const __bf16* bp = BT + (size_t)(n0 + r) * K + hf * 8;
    const size_t strA = (size_t)16 * K;

    // prologue
    v16bf a0 = frag_ld(ap);
    v16bf a1 = frag_ld(ap + strA);
    v16bf b0 = frag_ld(bp);
    v16bf b1 = frag_ld(bp + strA);

    for (int kc = 32; kc < K; kc += 32) {
        // prefetch next K-step while current WMMAs run
        v16bf na0 = frag_ld(ap + kc);
        v16bf na1 = frag_ld(ap + strA + kc);
        v16bf nb0 = frag_ld(bp + kc);
        v16bf nb1 = frag_ld(bp + strA + kc);
        acc[0][0] = wmma_bf16(a0, b0, acc[0][0]);
        acc[0][1] = wmma_bf16(a0, b1, acc[0][1]);
        acc[1][0] = wmma_bf16(a1, b0, acc[1][0]);
        acc[1][1] = wmma_bf16(a1, b1, acc[1][1]);
        a0 = na0; a1 = na1; b0 = nb0; b1 = nb1;
    }
    // epilogue K-step
    acc[0][0] = wmma_bf16(a0, b0, acc[0][0]);
    acc[0][1] = wmma_bf16(a0, b1, acc[0][1]);
    acc[1][0] = wmma_bf16(a1, b0, acc[1][0]);
    acc[1][1] = wmma_bf16(a1, b1, acc[1][1]);

#pragma unroll
    for (int i = 0; i < 2; ++i)
#pragma unroll
        for (int j = 0; j < 2; ++j) {
            int col = n0 + j * 16 + r;
            float bv = BIAS ? bias[col] : 0.0f;
#pragma unroll
            for (int e = 0; e < 8; ++e) {
                int row = m0 + i * 16 + e + 8 * hf;
                float v = acc[i][j][e];
                if (BIAS)  v += bv;
                if (RELU)  v = fmaxf(v, 0.0f);
                if (RESID) v += resid[(size_t)row * N + col];
                if (OUTF)  outf[(size_t)row * N + col] = v;
                if (OUTBF) outb[(size_t)row * N + col] = f2bf(v);
            }
        }
}

// ---------------------------------------------------------------------------
// 5. Transpose V per head:  v[B,L,D] -> vT[B,H,HD,L]  (bf16)
// ---------------------------------------------------------------------------
__global__ void vtrans_kernel(const __bf16* __restrict__ v, __bf16* __restrict__ vT) {
    __shared__ __bf16 tile[32][33];
    int bh = blockIdx.z;                       // b*H + h
    int b = bh / Hc, h = bh % Hc;
    int l0 = blockIdx.x * 32, d0 = blockIdx.y * 32;
    int tx = threadIdx.x, ty = threadIdx.y;    // 32 x 8
#pragma unroll
    for (int k = 0; k < 32; k += 8)
        tile[ty + k][tx] = v[((size_t)b * Lc + l0 + ty + k) * Dc + h * HDc + d0 + tx];
    __syncthreads();
#pragma unroll
    for (int k = 0; k < 32; k += 8)
        vT[((size_t)bh * HDc + d0 + ty + k) * Lc + l0 + tx] = tile[tx][ty + k];
}

// ---------------------------------------------------------------------------
// 6. BigBird attention: one WG per (query block, head, batch); 4 waves x 16 q
//    rows; flash-style online softmax over key blocks.  q pre-scaled 1/8.
//    amdgpu_num_vgpr(224) requests an explicit VGPR budget so the ~190-VGPR
//    live set (qa + scv + oacc + prefetched V fragments + softmax scalars)
//    is allocated spill-free instead of being folded to scratch by the
//    default 128-VGPR occupancy heuristic.
// ---------------------------------------------------------------------------
__global__ void __attribute__((amdgpu_num_vgpr(224)))
attn_kernel(const __bf16* __restrict__ q,
            const __bf16* __restrict__ k,
            const __bf16* __restrict__ vT,
            const int*    __restrict__ plan,
            __bf16* __restrict__ o) {
    int qb = blockIdx.x, h = blockIdx.y, b = blockIdx.z;
    int wave = threadIdx.x >> 5, lane = threadIdx.x & 31;
    int r = lane & 15, hf = lane >> 4;

    __shared__ __attribute__((aligned(16))) __bf16 Plds[4][16 * 64];
    __bf16* pl = &Plds[wave][0];

    int qrow0 = qb * 64 + wave * 16;
    const __bf16* qbase = q + ((size_t)b * Lc) * Dc + (size_t)h * HDc;
    v16bf qa0 = frag_ld(qbase + (size_t)(qrow0 + r) * Dc + 0  + hf * 8);
    v16bf qa1 = frag_ld(qbase + (size_t)(qrow0 + r) * Dc + 32 + hf * 8);

    bool isglobal = (qb == 0) || (qb == NBc - 1);
    int  nkb = isglobal ? NBc : 8;
    const int* prow = plan + (qb - 1) * 8;     // only read when !isglobal

    const __bf16* vhead = vT + ((size_t)(b * Hc + h)) * HDc * Lc;

    float m[8], l[8];
    v8f oacc[4];
#pragma unroll
    for (int e = 0; e < 8; ++e) { m[e] = -1e30f; l[e] = 0.0f; }
#pragma unroll
    for (int t = 0; t < 4; ++t) oacc[t] = v8f_zero();

    for (int s = 0; s < nkb; ++s) {
        int kb = isglobal ? s : prow[s];
        size_t koff = ((size_t)b * Lc + (size_t)kb * 64) * Dc + (size_t)h * HDc;

        // ---- S = q . k^T : K fragments pipelined one tile ahead ----
        const __bf16* kp = k + koff + (size_t)r * Dc + hf * 8;
        v16bf kf0 = frag_ld(kp);
        v16bf kf1 = frag_ld(kp + 32);
        v8f scv[4];
#pragma unroll
        for (int t = 0; t < 4; ++t) {
            v16bf nk0 = kf0, nk1 = kf1;
            if (t < 3) {
                const __bf16* kpn = kp + (size_t)(16 * (t + 1)) * Dc;
                nk0 = frag_ld(kpn);
                nk1 = frag_ld(kpn + 32);
            }
            v8f z = v8f_zero();
            z = wmma_bf16(qa0, kf0, z);
            scv[t] = wmma_bf16(qa1, kf1, z);
            kf0 = nk0; kf1 = nk1;
        }

        // ---- issue V kc=0 fragments now (hide behind softmax VALU) ----
        const __bf16* vp0 = vhead + (size_t)r * Lc + (size_t)kb * 64 + hf * 8;
        v16bf vf0[4];
#pragma unroll
        for (int t = 0; t < 4; ++t)
            vf0[t] = frag_ld(vp0 + (size_t)(16 * t) * Lc);

        // ---- row max (per element e: row = e + 8*hf) ----
        float mx[8];
#pragma unroll
        for (int e = 0; e < 8; ++e) {
            float v = fmaxf(fmaxf(scv[0][e], scv[1][e]), fmaxf(scv[2][e], scv[3][e]));
#pragma unroll
            for (int d = 8; d; d >>= 1) v = fmaxf(v, __shfl_xor(v, d));
            mx[e] = v;
        }
        float corr[8];
#pragma unroll
        for (int e = 0; e < 8; ++e) {
            float mn = fmaxf(m[e], mx[e]);
            corr[e] = __expf(m[e] - mn);
            m[e] = mn;
        }
        // ---- P = exp(S - m), row sums ----
#pragma unroll
        for (int t = 0; t < 4; ++t)
#pragma unroll
            for (int e = 0; e < 8; ++e) scv[t][e] = __expf(scv[t][e] - m[e]);
#pragma unroll
        for (int e = 0; e < 8; ++e) {
            float v = scv[0][e] + scv[1][e] + scv[2][e] + scv[3][e];
#pragma unroll
            for (int d = 8; d; d >>= 1) v += __shfl_xor(v, d);
            l[e] = l[e] * corr[e] + v;
        }
#pragma unroll
        for (int t = 0; t < 4; ++t)
#pragma unroll
            for (int e = 0; e < 8; ++e) oacc[t][e] *= corr[e];

        // ---- P: D-layout -> A-layout via wave-private LDS ----
#pragma unroll
        for (int t = 0; t < 4; ++t)
#pragma unroll
            for (int e = 0; e < 8; ++e)
                pl[(e + 8 * hf) * 64 + 16 * t + r] = f2bf(scv[t][e]);

        // ---- issue V kc=32 fragments; they fly during the LDS wait ----
        v16bf vf1[4];
#pragma unroll
        for (int t = 0; t < 4; ++t)
            vf1[t] = frag_ld(vp0 + (size_t)(16 * t) * Lc + 32);

        asm volatile("s_wait_dscnt 0" ::: "memory");   // cross-lane LDS visibility

        v16bf pa0 = frag_ld(pl + r * 64 + 0  + hf * 8);
        v16bf pa1 = frag_ld(pl + r * 64 + 32 + hf * 8);

        // ---- O += P . V ----
#pragma unroll
        for (int t = 0; t < 4; ++t)
            oacc[t] = wmma_bf16(pa0, vf0[t], oacc[t]);
#pragma unroll
        for (int t = 0; t < 4; ++t)
            oacc[t] = wmma_bf16(pa1, vf1[t], oacc[t]);

        asm volatile("s_wait_dscnt 0" ::: "memory");   // reads done before overwrite
    }

    // ---- finalize: O / l, store bf16 (B,L,D) ----
#pragma unroll
    for (int t = 0; t < 4; ++t)
#pragma unroll
        for (int e = 0; e < 8; ++e) {
            float ov = oacc[t][e] / l[e];
            o[((size_t)b * Lc + qrow0 + e + 8 * hf) * Dc + h * HDc + 16 * t + r] = f2bf(ov);
        }
}

// ---------------------------------------------------------------------------
// Launch
// ---------------------------------------------------------------------------
extern "C" void kernel_launch(void* const* d_in, const int* in_sizes, int n_in,
                              void* d_out, int out_size, void* d_ws, size_t ws_size,
                              hipStream_t stream) {
    const float* x    = (const float*)d_in[0];
    const float* ln1s = (const float*)d_in[1];
    const float* ln1b = (const float*)d_in[2];
    const float* Wq   = (const float*)d_in[3];
    const float* Wk   = (const float*)d_in[4];
    const float* Wv   = (const float*)d_in[5];
    const float* Wo   = (const float*)d_in[6];
    const float* ln2s = (const float*)d_in[7];
    const float* ln2b = (const float*)d_in[8];
    const float* W1   = (const float*)d_in[9];
    const float* b1   = (const float*)d_in[10];
    const float* W2   = (const float*)d_in[11];
    const float* b2   = (const float*)d_in[12];
    float* out = (float*)d_out;

    const size_t SZ = (size_t)Mrows * Dc * sizeof(__bf16);   // one bf16 plane
    char* w = (char*)d_ws;
    __bf16* hbuf = (__bf16*)(w + 0 * SZ);
    __bf16* qbuf = (__bf16*)(w + 1 * SZ);
    __bf16* kbuf = (__bf16*)(w + 2 * SZ);
    __bf16* vbuf = (__bf16*)(w + 3 * SZ);
    __bf16* vTb  = (__bf16*)(w + 4 * SZ);
    __bf16* obuf = (__bf16*)(w + 5 * SZ);
    float*  x1   = (float*)(w + 6 * SZ);
    char* p = w + 6 * SZ + (size_t)Mrows * Dc * sizeof(float);
    __bf16* wqT = (__bf16*)p; p += (size_t)Dc * Dc * 2;
    __bf16* wkT = (__bf16*)p; p += (size_t)Dc * Dc * 2;
    __bf16* wvT = (__bf16*)p; p += (size_t)Dc * Dc * 2;
    __bf16* woT = (__bf16*)p; p += (size_t)Dc * Dc * 2;
    __bf16* w1T = (__bf16*)p; p += (size_t)MLPc * Dc * 2;
    __bf16* w2T = (__bf16*)p; p += (size_t)Dc * MLPc * 2;
    int* plan   = (int*)p;
    __bf16* t1  = qbuf;   // MLP hidden aliases dead q..vT region (4*SZ = 50.3MB)
    __bf16* y   = hbuf;   // LN2 output aliases dead h

    dim3 tb(32, 8);

    // 1. plan
    plan_kernel<<<1, 32, 0, stream>>>(plan);
    // 2. weights -> transposed bf16 (q-scale 1/sqrt(64) folded into WqT)
    wtrans_kernel<<<dim3(Dc / 32, Dc / 32), tb, 0, stream>>>(Wq, wqT, Dc, Dc, 0.125f);
    wtrans_kernel<<<dim3(Dc / 32, Dc / 32), tb, 0, stream>>>(Wk, wkT, Dc, Dc, 1.0f);
    wtrans_kernel<<<dim3(Dc / 32, Dc / 32), tb, 0, stream>>>(Wv, wvT, Dc, Dc, 1.0f);
    wtrans_kernel<<<dim3(Dc / 32, Dc / 32), tb, 0, stream>>>(Wo, woT, Dc, Dc, 1.0f);
    wtrans_kernel<<<dim3(MLPc / 32, Dc / 32), tb, 0, stream>>>(W1, w1T, Dc, MLPc, 1.0f);
    wtrans_kernel<<<dim3(Dc / 32, MLPc / 32), tb, 0, stream>>>(W2, w2T, MLPc, Dc, 1.0f);
    // 3. LN1
    ln_kernel<<<Mrows, 256, 0, stream>>>(x, ln1s, ln1b, hbuf);
    // 4. QKV projections
    dim3 g768(Dc / 64, Mrows / 128);
    gemm_kernel<false, false, false, true, false><<<g768, 256, 0, stream>>>(
        hbuf, wqT, nullptr, nullptr, qbuf, nullptr, Mrows, Dc, Dc);
    gemm_kernel<false, false, false, true, false><<<g768, 256, 0, stream>>>(
        hbuf, wkT, nullptr, nullptr, kbuf, nullptr, Mrows, Dc, Dc);
    gemm_kernel<false, false, false, true, false><<<g768, 256, 0, stream>>>(
        hbuf, wvT, nullptr, nullptr, vbuf, nullptr, Mrows, Dc, Dc);
    // 5. V transpose per head
    vtrans_kernel<<<dim3(Lc / 32, HDc / 32, Bc * Hc), tb, 0, stream>>>(vbuf, vTb);
    // 6. attention
    attn_kernel<<<dim3(NBc, Hc, Bc), 128, 0, stream>>>(qbuf, kbuf, vTb, plan, obuf);
    // 7. output projection + residual -> x1 (f32)
    gemm_kernel<false, false, true, false, true><<<g768, 256, 0, stream>>>(
        obuf, woT, nullptr, x, nullptr, x1, Mrows, Dc, Dc);
    // 8. LN2
    ln_kernel<<<Mrows, 256, 0, stream>>>(x1, ln2s, ln2b, y);
    // 9. MLP up + bias + relu -> t1 (bf16)
    gemm_kernel<true, true, false, true, false><<<dim3(MLPc / 64, Mrows / 128), 256, 0, stream>>>(
        y, w1T, b1, nullptr, t1, nullptr, Mrows, MLPc, Dc);
    // 10. MLP down + bias + residual(x1) -> out (f32)
    gemm_kernel<true, false, true, false, true><<<g768, 256, 0, stream>>>(
        t1, w2T, b2, x1, nullptr, out, Mrows, Dc, MLPc);
}
